// PosDefInput_28776280883688
// MI455X (gfx1250) — matrix operand order repaired and checked by reading, not compiled, not run on hardware
//
#include <hip/hip_runtime.h>
#include <hip/hip_bf16.h>

typedef __attribute__((ext_vector_type(16))) _Float16 v16h;
typedef __attribute__((ext_vector_type(8)))  _Float16 v8h;
typedef __attribute__((ext_vector_type(8)))  float    v8f;
typedef __attribute__((ext_vector_type(4)))  float    v4f;
typedef __attribute__((ext_vector_type(4)))  unsigned int u32x4;
typedef __attribute__((ext_vector_type(8)))  int      i32x8;
typedef __attribute__((ext_vector_type(4)))  int      i32x4;

#define BT   128      // batch rows per workgroup
#define APAD 88       // x tile row stride (halves): 16B-aligned rows, conflict-free
#define HPAD 264      // h tile row stride (halves): 16B-aligned rows, conflict-free
#define XPAD 132      // xT row stride (floats): 16B-aligned rows

// -------- TDM availability (guarded; falls back to VGPR staging) -----------
#if defined(__has_builtin)
#if __has_builtin(__builtin_amdgcn_tensor_load_to_lds) && \
    __has_builtin(__builtin_amdgcn_s_wait_tensorcnt)
#define USE_TDM 1
#endif
#endif
#ifndef USE_TDM
#define USE_TDM 0
#endif

// ---------------------------------------------------------------------------
// Weight reformat: row-major f32 [K][N] -> f16 B-fragment-major layout.
// Fragment = one 32(K) x 16(N) tile in the contiguous-k B order used here (load_afrag uses the same k order):
//   lane l (0..31), element e (0..15):  K = kb*32 + e + 16*(l>>4), N = ct*16 + (l&15)
// Fragments ordered: for ct (16-col tile): for kb (32-K tile).
// ---------------------------------------------------------------------------
__global__ void prep_B_frags(const float* __restrict__ src, _Float16* __restrict__ dst,
                             int K, int N, int NKB) {
  int o = blockIdx.x * 256 + threadIdx.x;
  if (o >= K * N) return;
  int frag = o >> 9, fl = o & 511;
  int lane = fl >> 4, e = fl & 15;
  int ct = frag / NKB, kb = frag % NKB;
  int g = lane >> 4;
  int n = ct * 16 + (lane & 15);
  int k = kb * 32 + e + 16 * g;
  dst[o] = (_Float16)src[(size_t)k * N + n];
}

// W3 [256][2080] -> padded+reordered f16 fragments for the "Lfull" GEMM:
// column-tile t = i*4 + jt holds columns (i, j=jt*16..jt*16+15); zero where j>i.
__global__ void prep_W3_frags(const float* __restrict__ W3, _Float16* __restrict__ dst) {
  int o = blockIdx.x * 256 + threadIdx.x;             // 1,048,576 total
  int frag = o >> 9, fl = o & 511;
  int lane = fl >> 4, e = fl & 15;
  int t = frag >> 3, kb = frag & 7;
  int i = t >> 2, jt = t & 3;
  int g = lane >> 4;
  int j = jt * 16 + (lane & 15);
  int k = kb * 32 + e + 16 * g;
  float v = (j <= i) ? W3[(size_t)k * 2080 + (i * (i + 1) / 2 + j)] : 0.0f;
  dst[o] = (_Float16)v;
}

// b3 [2080] -> padded b3r[64][64] (zero above diagonal)
__global__ void prep_b3r(const float* __restrict__ b3, float* __restrict__ dst) {
  int idx = blockIdx.x * 256 + threadIdx.x;           // 4096
  int i = idx >> 6, j = idx & 63;
  dst[idx] = (j <= i) ? b3[i * (i + 1) / 2 + j] : 0.0f;
}

// ---------------------------------------------------------------------------
// Device helpers
// ---------------------------------------------------------------------------
__device__ __forceinline__ v8f wmma_f16(v16h a, v16h b, v8f c) {
  return __builtin_amdgcn_wmma_f32_16x16x32_f16(false, a, false, b, (short)0, c,
                                                false, false);
}

__device__ __forceinline__ v16h load_afrag(const _Float16* base, int stride, int M,
                                           int kb, int g) {
  const _Float16* p = base + M * stride + kb * 32 + 16 * g;
  v8h lo = *(const v8h*)(p);
  v8h hi = *(const v8h*)(p + 8);
  return __builtin_shufflevector(lo, hi, 0, 1, 2, 3, 4, 5, 6, 7,
                                 8, 9, 10, 11, 12, 13, 14, 15);
}

#if USE_TDM
__device__ __forceinline__ unsigned lds_addr_of(void* p) {
  return (unsigned)(unsigned long long)(__attribute__((address_space(3))) void*)p;
}

// One TDM descriptor: contiguous 8KB (1024 x 8B) global -> LDS.
__device__ __forceinline__ void tdm_load_8kb(unsigned lds_byte_addr, const void* gsrc) {
  unsigned long long ga = (unsigned long long)gsrc;
  u32x4 g0;
  g0.x = 1u;                                           // count=1, user D#
  g0.y = lds_byte_addr;                                // LDS dest (bytes)
  g0.z = (unsigned)(ga & 0xFFFFFFFFu);                 // global_addr[31:0]
  g0.w = (unsigned)((ga >> 32) & 0x01FFFFFFu)          // global_addr[56:32]
         | (2u << 30);                                 // type=2 ("image")
  i32x8 g1;
  g1[0] = 0x00030000;                                  // data_size=3 (8B), mask=0
  g1[1] = (int)(1024u << 16);                          // tensor_dim0[15:0]=1024
  g1[2] = (int)(1u << 16);                             // tensor_dim0 hi=0; tensor_dim1=1
  g1[3] = (int)(1024u << 16);                          // tensor_dim1 hi=0; tile_dim0=1024
  g1[4] = 1;                                           // tile_dim1=1, tile_dim2=0
  g1[5] = 1024;                                        // tensor_dim0_stride lo
  g1[6] = (int)(1024u << 16);                          // stride hi=0; dim1_stride lo=1024
  g1[7] = 0;
  i32x4 z4 = {0, 0, 0, 0};
  i32x8 z8 = {0, 0, 0, 0, 0, 0, 0, 0};
  __builtin_amdgcn_tensor_load_to_lds(g0, g1, z4, z4, z8, 0);
}
#endif

// Stage one 8KB B-tile (8 fragments) into LDS.
__device__ __forceinline__ void stage_tile(_Float16* dst, const _Float16* src,
                                           int tid, int w) {
#if USE_TDM
  if (w == 0) tdm_load_8kb(lds_addr_of(dst), src);     // one DMA per tile
  (void)tid;
#else
  *(v16h*)(dst + tid * 16) = *(const v16h*)(src + tid * 16);
  (void)w;
#endif
}

__device__ __forceinline__ void stage_wait() {
#if USE_TDM
  __builtin_amdgcn_s_wait_tensorcnt((short)0);         // no-op for non-issuing waves
#endif
}

// D-tile (+bias, ELU) -> f16 LDS h tile. M = w*16 + r + 8g, col = nb*16 + n.
__device__ __forceinline__ void store_h(_Float16* h, int w, int g, int n, int nb,
                                        v8f c, float bias) {
#pragma unroll
  for (int r = 0; r < 8; ++r) {
    float v = c[r] + bias;
    v = v > 0.0f ? v : (__expf(v) - 1.0f);            // ELU (alpha=1)
    h[(w * 16 + r + 8 * g) * HPAD + nb * 16 + n] = (_Float16)v;
  }
}

// ---------------------------------------------------------------------------
// Fused kernel: h1=elu(xW1+b1); h2=elu(h1W2+b2); y[b,j]=sum_i (h2*W3r+b3r)*x[b,i];
// out[b] = -sum_j y^2 + 1000*voff.   One WG = 128 rows, 8 waves (one M-tile each).
// ---------------------------------------------------------------------------
__global__ __launch_bounds__(256)
void fused_posdef(const float* __restrict__ x,
                  const float* __restrict__ b1,
                  const float* __restrict__ b2,
                  const _Float16* __restrict__ W1f,
                  const _Float16* __restrict__ W2f,
                  const _Float16* __restrict__ W3f,
                  const float* __restrict__ b3r,
                  const float* __restrict__ voff,
                  float* __restrict__ out) {
  __shared__ __align__(32) _Float16 hA[BT * APAD];    // x f16, A layout (22.5 KB)
  __shared__ __align__(32) _Float16 h1s[BT * HPAD];   // 66 KB
  __shared__ __align__(32) _Float16 h2s[BT * HPAD];   // 66 KB
  __shared__ __align__(32) float    xT[64 * XPAD];    // x f32 transposed (33 KB)
  __shared__ __align__(32) _Float16 Bst[2 * 4096];    // B double-buffer (16 KB)

  const int tid  = threadIdx.x;
  const int lane = tid & 31;
  const int w    = tid >> 5;        // wave = M-tile 0..7
  const int g    = lane >> 4;       // half-wave
  const int n    = lane & 15;       // N within tile
  const int bbase = blockIdx.x * BT;

  // ---- phase 0: load x tile (f16 A-layout + f32 transposed) ----
  for (int idx = tid; idx < BT * 64; idx += 256) {
    int b = idx >> 6, k = idx & 63;
    float v = x[(size_t)(bbase + b) * 64 + k];
    hA[b * APAD + k] = (_Float16)v;
    xT[k * XPAD + b] = v;
  }
  __syncthreads();

  const int M0 = w * 16 + n;        // this lane's A-matrix row

  // ---- GEMM1: h1 = elu(x @ W1 + b1), K=64, B streamed from L2 ----
  {
    v16h a0 = load_afrag(hA, APAD, M0, 0, g);
    v16h a1 = load_afrag(hA, APAD, M0, 1, g);
#pragma unroll 4
    for (int nb = 0; nb < 16; ++nb) {
      v16h bf0 = *(const v16h*)(W1f + (size_t)(nb * 2 + 0) * 512 + lane * 16);
      v16h bf1 = *(const v16h*)(W1f + (size_t)(nb * 2 + 1) * 512 + lane * 16);
      v8f c = {};
      c = wmma_f16(a0, bf0, c);
      c = wmma_f16(a1, bf1, c);
      store_h(h1s, w, g, n, nb, c, b1[nb * 16 + n]);
    }
  }

  // ---- GEMM2: h2 = elu(h1 @ W2 + b2), K=256, B double-buffered via LDS ----
  v16h af[8];
#pragma unroll
  for (int kb = 0; kb < 8; ++kb) af[kb] = load_afrag(h1s, HPAD, M0, kb, g);

  stage_tile(Bst, W2f, tid, w);
  stage_wait();
  __syncthreads();
  for (int nb = 0; nb < 16; ++nb) {
    int buf = nb & 1;
    if (nb + 1 < 16) stage_tile(Bst + (buf ^ 1) * 4096, W2f + (size_t)(nb + 1) * 4096, tid, w);
    v8f c = {};
#pragma unroll
    for (int kb = 0; kb < 8; ++kb) {
      v16h bf = *(const v16h*)(Bst + buf * 4096 + kb * 512 + lane * 16);
      c = wmma_f16(af[kb], bf, c);
    }
    store_h(h2s, w, g, n, nb, c, b2[nb * 16 + n]);
    stage_wait();
    __syncthreads();
  }

  // ---- GEMM3 + fused tril reduction ----
#pragma unroll
  for (int kb = 0; kb < 8; ++kb) af[kb] = load_afrag(h2s, HPAD, M0, kb, g);

  float q[8];
#pragma unroll
  for (int r = 0; r < 8; ++r) q[r] = 0.0f;

  stage_tile(Bst, W3f, tid, w);     // first tile: (i=0, jt=0) = tile 0
  stage_wait();
  __syncthreads();
  int buf = 0;
  for (int jt = 0; jt < 4; ++jt) {
    float yacc[8];
#pragma unroll
    for (int r = 0; r < 8; ++r) yacc[r] = 0.0f;
    for (int i = jt * 16; i < 64; ++i) {
      // prefetch next non-zero tile
      int ni = i + 1, njt = jt;
      if (ni == 64) { njt = jt + 1; ni = njt * 16; }
      if (njt < 4)
        stage_tile(Bst + (buf ^ 1) * 4096, W3f + (size_t)(ni * 4 + njt) * 4096, tid, w);

      v8f c = {};
#pragma unroll
      for (int kb = 0; kb < 8; ++kb) {
        v16h bf = *(const v16h*)(Bst + buf * 4096 + kb * 512 + lane * 16);
        c = wmma_f16(af[kb], bf, c);
      }
      float bias = b3r[i * 64 + jt * 16 + n];         // 0 above diagonal
      const float* xp = xT + i * XPAD + w * 16 + 8 * g;
      v4f x0 = *(const v4f*)(xp);
      v4f x1 = *(const v4f*)(xp + 4);
#pragma unroll
      for (int r = 0; r < 4; ++r) yacc[r]     += (c[r]     + bias) * x0[r];
#pragma unroll
      for (int r = 0; r < 4; ++r) yacc[r + 4] += (c[r + 4] + bias) * x1[r];
      stage_wait();
      __syncthreads();
      buf ^= 1;
    }
#pragma unroll
    for (int r = 0; r < 8; ++r) q[r] += yacc[r] * yacc[r];
  }

  // ---- reduce q over the 16 lanes of each half-wave (j dimension) ----
  float off = voff[0] * 1000.0f;
#pragma unroll
  for (int r = 0; r < 8; ++r) {
    float v = q[r];
    v += __shfl_xor(v, 1);
    v += __shfl_xor(v, 2);
    v += __shfl_xor(v, 4);
    v += __shfl_xor(v, 8);
    q[r] = v;
  }
  if (n == 0) {
#pragma unroll
    for (int r = 0; r < 8; ++r)
      out[bbase + w * 16 + g * 8 + r] = -q[r] + off;
  }
}

// ---------------------------------------------------------------------------
extern "C" void kernel_launch(void* const* d_in, const int* in_sizes, int n_in,
                              void* d_out, int out_size, void* d_ws, size_t ws_size,
                              hipStream_t stream) {
  const float* x    = (const float*)d_in[0];
  const float* W1   = (const float*)d_in[1];
  const float* b1   = (const float*)d_in[2];
  const float* W2   = (const float*)d_in[3];
  const float* b2   = (const float*)d_in[4];
  const float* W3   = (const float*)d_in[5];
  const float* b3   = (const float*)d_in[6];
  const float* voff = (const float*)d_in[7];
  float* out = (float*)d_out;

  char* ws = (char*)d_ws;
  _Float16* W1f = (_Float16*)(ws);                               //  32 KB
  _Float16* W2f = (_Float16*)(ws + 32768);                       // 128 KB
  _Float16* W3f = (_Float16*)(ws + 32768 + 131072);              //   2 MB
  float*    b3r = (float*)   (ws + 32768 + 131072 + 2097152);    //  16 KB

  prep_B_frags<<<64,   256, 0, stream>>>(W1, W1f, 64, 256, 2);
  prep_B_frags<<<256,  256, 0, stream>>>(W2, W2f, 256, 256, 8);
  prep_W3_frags<<<4096, 256, 0, stream>>>(W3, W3f);
  prep_b3r<<<16, 256, 0, stream>>>(b3, b3r);

  fused_posdef<<<32768 / BT, 256, 0, stream>>>(x, b1, b2, W1f, W2f, W3f, b3r,
                                               voff, out);
}
